// JAX_GeomAttention_15479062135061
// MI455X (gfx1250) — compile-verified, matrix operand order checked by reading, not run on hardware
//
#include <hip/hip_runtime.h>
#include <hip/hip_bf16.h>
#include <math.h>

// ---------------------------------------------------------------------------
// Geometric attention for MI455X (gfx1250, wave32, WMMA bf16 path).
//   scores[l,s] = ((1-a)*dot + a*sqrt(relu(q2[l]*k2[l] - dot^2) + 1e-8)) / 8
//   (k2 indexed by l, reproducing the reference broadcast faithfully)
//   out = softmax_s(scores) @ V   ;  plus scalar mean(|scores|)
//
// Tiling: 128 Q-rows per block (4 waves x 32 rows), S consumed in 32-column
// tiles. Each wave runs 16 v_wmma_f32_16x16x32_bf16 per S-tile; every K/V
// B-fragment loaded from LDS feeds two WMMAs (two 16-row sub-tiles).
// The next tile's K/V rows are pulled into cache with global_prefetch_b8
// (no VGPR / LOADcnt cost) while the current tile's WMMAs run, keeping the
// kernel under the 256-VGPR window (no s_set_vgpr_msb churn).
// ---------------------------------------------------------------------------

typedef __attribute__((ext_vector_type(16))) __bf16 v16bf;
typedef __attribute__((ext_vector_type(8)))  float  v8f;

static constexpr int BB = 4;
static constexpr int LL = 2048;
static constexpr int HH = 8;
static constexpr int EE = 64;
static constexpr int SS = 2048;
static constexpr float ALPHA = 0.5f;
static constexpr int MB = 128;     // Q rows per block
static constexpr int MW = 32;      // Q rows per wave (2 x 16-row sub-tiles)

// Load a 16-bit WMMA operand fragment (16 K-values per lane) from an LDS row.
// Packing follows the CDNA5 ISA 16-bit A/B layout: lanes 0-15 carry K = kc..kc+7
// (v0..v3) and K = kc+16..kc+23 (v4..v7); lanes 16-31 carry the +8 halves.
__device__ __forceinline__ v16bf load_frag16(const __bf16* rowbase, int kc32, int hf) {
    v16bf a;
    const __bf16* p0 = rowbase + kc32 + hf * 8;
    const __bf16* p1 = rowbase + kc32 + 16 + hf * 8;
#pragma unroll
    for (int i = 0; i < 8; ++i) {
        a[i]     = p0[i];
        a[8 + i] = p1[i];
    }
    return a;
}

__global__ void __launch_bounds__(128)
geom_attn_zero_scalar(float* p) {
    if (threadIdx.x == 0 && blockIdx.x == 0) *p = 0.0f;
}

__global__ void __launch_bounds__(128)
geom_attn_kernel(const float* __restrict__ Q,
                 const float* __restrict__ K,
                 const float* __restrict__ V,
                 float* __restrict__ Out,
                 float* __restrict__ absOut) {
    // Block decode: blockIdx.x = (b*HH + h)*(LL/MB) + ltile
    const int ltiles = LL / MB;
    const int ltile  = blockIdx.x % ltiles;
    const int h      = (blockIdx.x / ltiles) % HH;
    const int b      = blockIdx.x / (ltiles * HH);
    const int l0     = ltile * MB;

    const int tid  = threadIdx.x;
    const int wave = tid >> 5;
    const int lane = tid & 31;
    const int lm   = lane & 15;   // column / row-in-half index
    const int hf   = lane >> 4;   // half-wave select

    __shared__ __bf16 sQ[MB * 64];       // Q tile, row-major bf16       (16 KB)
    __shared__ __bf16 sK[32 * 64];       // K s-tile, row-major bf16      (4 KB)
    __shared__ __bf16 sVT[64 * 32];      // V s-tile, e-major (transposed)(4 KB)
    __shared__ __bf16 sP[4][MW * 32];    // per-wave softmax P tiles      (8 KB)
    __shared__ float  sQ2K2[MB];         // q2[l]*k2[l] per Q row
    __shared__ float  sRed[4];

    // ---- Stage Q tile (fp32 -> bf16) ----------------------------------
    for (int u = tid; u < MB * 16; u += 128) {           // float4 units
        const int r  = u >> 4;
        const int c4 = (u & 15) << 2;
        const float4 q4 = *reinterpret_cast<const float4*>(
            &Q[((size_t)((b * LL + l0 + r) * HH + h)) * EE + c4]);
        __bf16* d = &sQ[r * 64 + c4];
        d[0] = (__bf16)q4.x; d[1] = (__bf16)q4.y;
        d[2] = (__bf16)q4.z; d[3] = (__bf16)q4.w;
    }
    // ---- Per-row q2[l]*k2[l] (same row index l for both: faithful bug) -
    {
        const int l = l0 + tid;              // MB == blockDim.x == 128
        const float* qr = &Q[((size_t)((b * LL + l) * HH + h)) * EE];
        const float* kr = &K[((size_t)((b * LL + l) * HH + h)) * EE];
        float q2 = 0.f, k2 = 0.f;
#pragma unroll 8
        for (int e = 0; e < EE; ++e) { q2 += qr[e] * qr[e]; k2 += kr[e] * kr[e]; }
        sQ2K2[tid] = q2 * k2;
    }
    __syncthreads();

    // ---- Resident per-wave state (two 16-row sub-tiles r = 0,1) -------
    v16bf aQ[2][2];
#pragma unroll
    for (int r = 0; r < 2; ++r)
#pragma unroll
        for (int kc = 0; kc < 2; ++kc)
            aQ[r][kc] = load_frag16(&sQ[(wave * MW + r * 16 + lm) * 64], kc * 32, hf);

    float q2k2r[2][8];
#pragma unroll
    for (int r = 0; r < 2; ++r)
#pragma unroll
        for (int j = 0; j < 8; ++j)   // C-layout row M = j + 8*hf
            q2k2r[r][j] = sQ2K2[wave * MW + r * 16 + hf * 8 + j];

    float mrow[2][8], lrow[2][8];
#pragma unroll
    for (int r = 0; r < 2; ++r)
#pragma unroll
        for (int j = 0; j < 8; ++j) { mrow[r][j] = -1e30f; lrow[r][j] = 0.f; }
    v8f oacc[2][4] = {};        // 2 x (16x64) f32 output accumulators
    float absSum = 0.f;

    const float scale = 0.125f;             // 1/sqrt(64)

    // ---- Main loop over key/value tiles (32 columns each) -------------
    for (int s0 = 0; s0 < SS; s0 += 32) {
        // Prefetch next tile's K/V rows into cache (global_prefetch_b8:
        // no VGPR, no LOADcnt). 64 threads cover 32 K-rows + 32 V-rows;
        // each 256 B row = 2 cacheline-granular prefetches.
        if (s0 + 32 < SS && tid < 64) {
            const int r = tid & 31;
            const float* base = (tid < 32 ? K : V)
                + ((size_t)((b * SS + s0 + 32 + r) * HH + h)) * EE;
            __builtin_prefetch(base, 0, 1);
            __builtin_prefetch(base + 32, 0, 1);
        }

        // Cooperative stage: K row-major, V transposed, fp32 -> bf16
        for (int u = tid; u < 32 * 16; u += 128) {
            const int r  = u >> 4;
            const int c4 = (u & 15) << 2;
            const size_t g = ((size_t)((b * SS + s0 + r) * HH + h)) * EE + c4;
            const float4 kk = *reinterpret_cast<const float4*>(&K[g]);
            const float4 vv = *reinterpret_cast<const float4*>(&V[g]);
            __bf16* kd = &sK[r * 64 + c4];
            kd[0] = (__bf16)kk.x; kd[1] = (__bf16)kk.y;
            kd[2] = (__bf16)kk.z; kd[3] = (__bf16)kk.w;
            sVT[(c4 + 0) * 32 + r] = (__bf16)vv.x;
            sVT[(c4 + 1) * 32 + r] = (__bf16)vv.y;
            sVT[(c4 + 2) * 32 + r] = (__bf16)vv.z;
            sVT[(c4 + 3) * 32 + r] = (__bf16)vv.w;
        }
        __syncthreads();

        // K^T B-fragments: loaded once, reused by both row sub-tiles
        v16bf bK[2][2];
#pragma unroll
        for (int t = 0; t < 2; ++t) {
            const __bf16* krow = &sK[(t * 16 + lm) * 64];
            bK[t][0] = load_frag16(krow, 0, hf);
            bK[t][1] = load_frag16(krow, 32, hf);
        }

        v16bf pA[2];
#pragma unroll
        for (int r = 0; r < 2; ++r) {
            // ---- scores = Q · K^T (two 16-column N-tiles) -------------
            float sc[2][8];
#pragma unroll
            for (int t = 0; t < 2; ++t) {
                v8f c = {};
                c = __builtin_amdgcn_wmma_f32_16x16x32_bf16(
                        false, aQ[r][0], false, bK[t][0], (short)0, c, false, false);
                c = __builtin_amdgcn_wmma_f32_16x16x32_bf16(
                        false, aQ[r][1], false, bK[t][1], (short)0, c, false, false);
#pragma unroll
                for (int j = 0; j < 8; ++j) {
                    const float d = c[j];
                    const float w = fmaxf(q2k2r[r][j] - d * d, 0.f);
                    const float s = ((1.f - ALPHA) * d
                                     + ALPHA * __builtin_amdgcn_sqrtf(w + 1e-8f)) * scale;
                    sc[t][j] = s;
                    absSum += fabsf(s);
                }
            }

            // ---- Online softmax (row-wise across the 16-lane half) ----
#pragma unroll
            for (int j = 0; j < 8; ++j) {
                float mx = fmaxf(sc[0][j], sc[1][j]);
#pragma unroll
                for (int off = 1; off < 16; off <<= 1)
                    mx = fmaxf(mx, __shfl_xor(mx, off, 32));
                const float mnew = fmaxf(mrow[r][j], mx);
                const float corr = __expf(mrow[r][j] - mnew);
                const float p0 = __expf(sc[0][j] - mnew);
                const float p1 = __expf(sc[1][j] - mnew);
                float psum = p0 + p1;
#pragma unroll
                for (int off = 1; off < 16; off <<= 1)
                    psum += __shfl_xor(psum, off, 32);
                lrow[r][j] = lrow[r][j] * corr + psum;
                mrow[r][j] = mnew;
#pragma unroll
                for (int n = 0; n < 4; ++n) oacc[r][n][j] *= corr;
                const int row = r * 16 + hf * 8 + j;          // C-layout row
                sP[wave][row * 32 + lm]      = (__bf16)p0;
                sP[wave][row * 32 + 16 + lm] = (__bf16)p1;    // wave-private LDS
            }
            pA[r] = load_frag16(&sP[wave][(r * 16 + lm) * 32], 0, hf);
        }

        // ---- O += P · V  (K = 32 over the s-tile) ---------------------
#pragma unroll
        for (int n = 0; n < 4; ++n) {
            const v16bf vB = load_frag16(&sVT[(n * 16 + lm) * 32], 0, hf);
#pragma unroll
            for (int r = 0; r < 2; ++r)
                oacc[r][n] = __builtin_amdgcn_wmma_f32_16x16x32_bf16(
                    false, pA[r], false, vB, (short)0, oacc[r][n], false, false);
        }
        __syncthreads();   // protect sK / sVT before next stage
    }

    // ---- Normalize and write output [B, L, H, E] ----------------------
#pragma unroll
    for (int r = 0; r < 2; ++r) {
        float invl[8];
#pragma unroll
        for (int j = 0; j < 8; ++j) invl[j] = __builtin_amdgcn_rcpf(lrow[r][j]);
#pragma unroll
        for (int n = 0; n < 4; ++n) {
#pragma unroll
            for (int j = 0; j < 8; ++j) {
                const int l = l0 + wave * MW + r * 16 + hf * 8 + j;
                const int e = n * 16 + lm;
                Out[((size_t)((b * LL + l) * HH + h)) * EE + e] = oacc[r][n][j] * invl[j];
            }
        }
    }

    // ---- mean(|scores|) reduction -------------------------------------
#pragma unroll
    for (int off = 1; off < 32; off <<= 1)
        absSum += __shfl_xor(absSum, off, 32);
    if (lane == 0) sRed[wave] = absSum;
    __syncthreads();
    if (tid == 0) {
        const float tot = sRed[0] + sRed[1] + sRed[2] + sRed[3];
        atomicAdd(absOut, tot * (1.0f / ((float)BB * HH * (float)LL * (float)SS)));
    }
}

extern "C" void kernel_launch(void* const* d_in, const int* in_sizes, int n_in,
                              void* d_out, int out_size, void* d_ws, size_t ws_size,
                              hipStream_t stream) {
    (void)in_sizes; (void)n_in; (void)d_ws; (void)ws_size;
    const float* Q = (const float*)d_in[0];
    const float* K = (const float*)d_in[1];
    const float* V = (const float*)d_in[2];
    float* Out    = (float*)d_out;
    float* absOut = Out + (out_size - 1);

    geom_attn_zero_scalar<<<1, 128, 0, stream>>>(absOut);
    const int blocks = BB * HH * (LL / MB);   // 512
    geom_attn_kernel<<<blocks, 128, 0, stream>>>(Q, K, V, Out, absOut);
}
